// ConvBlock_88167088652501
// MI455X (gfx1250) — matrix operand order repaired and checked by reading, not compiled, not run on hardware
//
#include <hip/hip_runtime.h>

// ---------------------------------------------------------------------------
// KPConv block (conv + group-norm + leaky-relu) for MI455X / gfx1250.
// M=32768 queries, H=32 neighbors, K=15 kernel points, Cin=Cout=64.
// Strategy:
//   stage 1 (per query):  weighted[k,c] = sum_h w[h,k]*F[h,c]
//       -> v_wmma_f32_16x16x32_bf16, A = w^T (16x32), B = F (32x16 chunks)
//   stage 2 (per 16-query tile): out[m,d] = sum_kc weighted[m,kc]*W[kc,d]
//       -> [16 x 960] x [960 x 64] GEMM, 30 WMMA k-steps per wave
//   group-norm: per-wave shfl reduction -> global float atomics -> finalize
//   kernel -> elementwise normalize + leaky relu.
// ---------------------------------------------------------------------------

typedef float  v8f   __attribute__((ext_vector_type(8)));
typedef __bf16 v16bf __attribute__((ext_vector_type(16)));
typedef __bf16 v8bf  __attribute__((ext_vector_type(8)));

union V16 { v16bf v; v8bf h[2]; };

#define NPTS   32768
#define MQ     32768
#define HNBR   32
#define KPTS   15
#define CIN    64
#define COUT   64
#define KC     (KPTS * CIN)        // 960 = 30 * 32, exact multiple of WMMA K
#define RS     976                 // padded row stride (elements) for weighted tile
#define INV_SIGMA 10.0f

static __device__ inline v8f wmma_bf16(v16bf a, v16bf b, v8f c) {
  // 8 args: (neg_a, A, neg_b, B, c_mod, C, reuse_a, reuse_b)
  return __builtin_amdgcn_wmma_f32_16x16x32_bf16(false, a, false, b, (short)0, c,
                                                 false, false);
}

// --- kernel 0: transpose+convert weights [15][64][64] f32 -> Wt[64][960] bf16,
//               and zero the 32-float stats region (graph-safe re-init). -----
__global__ void prep_kernel(const float* __restrict__ w_in,
                            __bf16* __restrict__ Wt,
                            float* __restrict__ stats) {
  int i = blockIdx.x * blockDim.x + threadIdx.x;
  if (i < KC * COUT) {
    int d  = i & 63;          // cout
    int kc = i >> 6;          // k*64 + c
    Wt[(size_t)d * KC + kc] = (__bf16)w_in[i];
  }
  if (blockIdx.x == 0 && threadIdx.x < 32) stats[threadIdx.x] = 0.f;
}

// --- kernel 1: main KPConv. 128 threads (4 waves), 16 queries per block. ----
__global__ __launch_bounds__(128, 1)
void kpconv_kernel(const float* __restrict__ s_feats,
                   const float* __restrict__ q_points,
                   const float* __restrict__ s_points,
                   const int*   __restrict__ nbr_idx,
                   const float* __restrict__ kpts,
                   const __bf16* __restrict__ Wt,
                   const float* __restrict__ bias,
                   float* __restrict__ y_out,
                   float* __restrict__ stats) {
  __shared__ __attribute__((aligned(16))) __bf16 lds_weighted[16 * RS];   // 31232 B
  __shared__ __attribute__((aligned(16))) __bf16 lds_F[4][CIN * HNBR];    // 16384 B  [c][h]
  __shared__ __attribute__((aligned(16))) float  lds_rel[4][HNBR * 4];    //  2048 B
  __shared__ float lds_inv[16];

  const int wv   = threadIdx.x >> 5;    // wave id 0..3
  const int lane = threadIdx.x & 31;
  const int hi   = lane >> 4;           // lane-half select
  const int lo   = lane & 15;
  const int q0   = blockIdx.x * 16;     // first query of this tile

  // kernel point assigned to this lane's A-matrix row (k = lo; row 15 is zero pad)
  float kx = 0.f, ky = 0.f, kz = 0.f;
  if (lo < KPTS) { kx = kpts[lo * 3 + 0]; ky = kpts[lo * 3 + 1]; kz = kpts[lo * 3 + 2]; }

  for (int r = 0; r < 4; ++r) {
    const int ql = r * 4 + wv;          // local query 0..15
    const int qg = q0 + ql;

    // ---- gather: lane = neighbor h ------------------------------------
    const int  idx = nbr_idx[qg * HNBR + lane];
    const bool vld = ((unsigned)idx < (unsigned)NPTS);
    const float qx = q_points[qg * 3 + 0];
    const float qy = q_points[qg * 3 + 1];
    const float qz = q_points[qg * 3 + 2];
    float px = 1e6f, py = 1e6f, pz = 1e6f;
    if (vld) { px = s_points[idx * 3 + 0]; py = s_points[idx * 3 + 1]; pz = s_points[idx * 3 + 2]; }
    lds_rel[wv][lane * 4 + 0] = px - qx;
    lds_rel[wv][lane * 4 + 1] = py - qy;
    lds_rel[wv][lane * 4 + 2] = pz - qz;
    lds_rel[wv][lane * 4 + 3] = 0.f;

    float fsum = 0.f;
    const float4* frow = (const float4*)(s_feats + (size_t)(vld ? idx : 0) * CIN);
    #pragma unroll
    for (int c = 0; c < CIN; c += 4) {
      float4 v = frow[c >> 2];
      if (!vld) { v.x = 0.f; v.y = 0.f; v.z = 0.f; v.w = 0.f; }
      lds_F[wv][(c + 0) * HNBR + lane] = (__bf16)v.x;   // transposed [c][h]
      lds_F[wv][(c + 1) * HNBR + lane] = (__bf16)v.y;
      lds_F[wv][(c + 2) * HNBR + lane] = (__bf16)v.z;
      lds_F[wv][(c + 3) * HNBR + lane] = (__bf16)v.w;
      fsum += v.x + v.y + v.z + v.w;
    }
    unsigned long long bm = __ballot(fsum > 0.f);
    int cnt = __popcll(bm);
    if (cnt < 1) cnt = 1;
    if (lane == 0) lds_inv[ql] = 1.f / (float)cnt;
    __syncthreads();

    // ---- stage-1 WMMA: A = w^T (16x32 bf16, row k = lo) ----------------
    V16 A;
    #pragma unroll
    for (int e = 0; e < 16; ++e) {
      const int h = hi * 8 + e + ((e >= 8) ? 8 : 0);     // A-layout K index
      float4 rel = *(const float4*)&lds_rel[wv][h * 4];
      const float dx = rel.x - kx, dy = rel.y - ky, dz = rel.z - kz;
      float w = 1.f - sqrtf(dx * dx + dy * dy + dz * dz) * INV_SIGMA;
      w = fmaxf(w, 0.f);
      if (lo == 15) w = 0.f;                             // zero pad row k=15
      A.v[e] = (__bf16)w;
    }
    #pragma unroll
    for (int cc = 0; cc < 4; ++cc) {
      const int c = cc * 16 + lo;                        // B column (N)
      V16 B;                                             // B-layout: K = 8*hi + e (+16 for upper half)
      B.h[0] = *(const v8bf*)&lds_F[wv][c * HNBR + hi * 8 + 0];
      B.h[1] = *(const v8bf*)&lds_F[wv][c * HNBR + hi * 8 + 16];
      v8f acc = {};
      acc = wmma_bf16(A.v, B.v, acc);
      #pragma unroll
      for (int rr = 0; rr < 8; ++rr) {
        const int k = rr + 8 * hi;                       // D row
        if (k < KPTS)
          lds_weighted[ql * RS + k * CIN + c] = (__bf16)acc[rr];
      }
    }
    __syncthreads();
  }

  // ---- stage-2 WMMA: [16 x 960] x [960 x 64] --------------------------
  const int d = wv * 16 + lo;                            // output channel (N)
  const float bias_d = bias[d];
  const __bf16* wtrow = Wt + (size_t)d * KC;
  v8f acc = {};
  for (int kc0 = 0; kc0 < KC; kc0 += 32) {
    __builtin_prefetch(wtrow + kc0 + 512, 0, 1);         // global_prefetch_b8
    V16 A2, B2;
    // A-layout: kc = kc0 + 8*hi + e (+16 for upper half)
    A2.h[0] = *(const v8bf*)&lds_weighted[lo * RS + kc0 + 8 * hi + 0];
    A2.h[1] = *(const v8bf*)&lds_weighted[lo * RS + kc0 + 8 * hi + 16];
    // B-layout: kc = kc0 + 8*hi + e (+16 for upper half)
    B2.h[0] = *(const v8bf*)(wtrow + kc0 + 8 * hi + 0);
    B2.h[1] = *(const v8bf*)(wtrow + kc0 + 8 * hi + 16);
    acc = wmma_bf16(A2.v, B2.v, acc);
  }

  // ---- epilogue: neighbor normalize + bias, write y, group-stat sums ---
  float gsum = 0.f, gsq = 0.f;
  #pragma unroll
  for (int rr = 0; rr < 8; ++rr) {
    const int ml = rr + 8 * hi;                          // local query (D row)
    const float y = acc[rr] * lds_inv[ml] + bias_d;
    y_out[(size_t)(q0 + ml) * COUT + d] = y;
    gsum += y; gsq += y * y;
  }
  // reduce across lanes sharing the same group (bit3 of lane fixed)
  #pragma unroll
  for (int m = 0; m < 4; ++m) {
    const int mask = (1 << m) == 8 ? 16 : (1 << m);      // 1,2,4,16
    gsum += __shfl_xor(gsum, mask, 32);
    gsq  += __shfl_xor(gsq,  mask, 32);
  }
  const int g = wv * 2 + ((lane >> 3) & 1);              // group = d>>3
  if (lane == 0 || lane == 8) {
    atomicAdd(&stats[g],     gsum);
    atomicAdd(&stats[8 + g], gsq);
  }
}

// --- kernel 2: finalize group stats ----------------------------------------
__global__ void finalize_kernel(const float* __restrict__ stats,
                                float* __restrict__ mean_rs) {
  int g = threadIdx.x;
  if (g < 8) {
    const float cnt  = (float)MQ * (COUT / 8);
    const float mean = stats[g] / cnt;
    const float var  = stats[8 + g] / cnt - mean * mean;
    mean_rs[g]     = mean;
    mean_rs[8 + g] = rsqrtf(var + 1e-5f);
  }
}

// --- kernel 3: apply group-norm + leaky relu, in place on d_out -------------
__global__ void gn_act_kernel(float* __restrict__ y,
                              const float* __restrict__ mean_rs,
                              const float* __restrict__ gamma,
                              const float* __restrict__ beta) {
  const int i = blockIdx.x * 256 + threadIdx.x;
  const int d = i & 63;
  const int g = d >> 3;
  const float x = y[i];
  const float t = (x - mean_rs[g]) * mean_rs[8 + g] * gamma[d] + beta[d];
  y[i] = (t >= 0.f) ? t : 0.1f * t;
}

extern "C" void kernel_launch(void* const* d_in, const int* in_sizes, int n_in,
                              void* d_out, int out_size, void* d_ws, size_t ws_size,
                              hipStream_t stream) {
  const float* s_feats  = (const float*)d_in[0];
  const float* q_points = (const float*)d_in[1];
  const float* s_points = (const float*)d_in[2];
  const int*   nbr_idx  = (const int*)  d_in[3];
  const float* kpts     = (const float*)d_in[4];
  const float* weights  = (const float*)d_in[5];
  const float* bias     = (const float*)d_in[6];
  const float* gamma    = (const float*)d_in[7];
  const float* beta     = (const float*)d_in[8];
  float* out = (float*)d_out;

  // workspace layout: [Wt bf16 64x960 = 122880 B][stats 16 f32][mean_rs 16 f32]
  __bf16* Wt    = (__bf16*)d_ws;
  float*  stats = (float*)((char*)d_ws + (size_t)KC * COUT * sizeof(__bf16));
  float*  mnrs  = stats + 16;

  prep_kernel<<<dim3((KC * COUT + 255) / 256), dim3(256), 0, stream>>>(weights, Wt, stats);
  kpconv_kernel<<<dim3(MQ / 16), dim3(128), 0, stream>>>(
      s_feats, q_points, s_points, nbr_idx, kpts, Wt, bias, out, stats);
  finalize_kernel<<<dim3(1), dim3(32), 0, stream>>>(stats, mnrs);
  gn_act_kernel<<<dim3(MQ * COUT / 256), dim3(256), 0, stream>>>(out, mnrs, gamma, beta);
}
